// BlockSparseFlashAttention_48086453846657
// MI455X (gfx1250) — matrix operand order repaired and checked by reading, not compiled, not run on hardware
//
#include <hip/hip_runtime.h>
#include <hip/hip_bf16.h>

typedef __attribute__((ext_vector_type(16))) _Float16 v16h;
typedef __attribute__((ext_vector_type(8)))  _Float16 v8h;
typedef __attribute__((ext_vector_type(8)))  float    v8f;
typedef __attribute__((ext_vector_type(4)))  float    v4f;

#define SEQ   2048
#define NH    32
#define NKV   8
#define HD    128
#define BLK   64
#define NB    (SEQ / BLK)          // 32
#define LOCAL 16
#define VERT  8
#define GQA   (NH / NKV)           // 4
#define ATT_SCALE 0.08838834764831845f

// One workgroup = one (query block qb, head h). 128 threads = 4 waves,
// wave w handles query rows [qb*64 + w*16, +16).
__global__ __launch_bounds__(128) void bsfa_kernel(
    const float* __restrict__ Q,   // [SEQ, NH*HD]
    const float* __restrict__ K,   // [SEQ, NKV*HD]
    const float* __restrict__ V,   // [SEQ, NKV*HD]
    float* __restrict__ O)         // [SEQ, NH*HD]
{
    __shared__ __align__(32) _Float16 Ks[BLK][HD];   // 16 KB, row-major (key, dim)
    __shared__ __align__(32) _Float16 Vt[HD][BLK];   // 16 KB, transposed (dim, key)
    __shared__ __align__(32) _Float16 Ps[4][16][BLK];// 8 KB, per-wave P scratch

    const int qb   = blockIdx.x;
    const int h    = blockIdx.y;
    const int hk   = h / GQA;
    const int tid  = threadIdx.x;
    const int wave = tid >> 5;
    const int lane = tid & 31;
    const int ln16 = lane & 15;
    const int hi   = lane >> 4;          // which half of the wave

    // ---- Q strip -> 4 f16 A-fragments (16 rows x 32 dims each) -------------
    // A layout (16-bit 16x32): lane(l<16)=row M, halves:
    //   j=0..15 -> k = k0 + hi*16 + j  (contiguous, matches B)
    v16h qfrag[4];
    {
        const int qrow = qb * BLK + wave * 16 + ln16;
        const float* qp = Q + (size_t)qrow * (NH * HD) + (size_t)h * HD;
        for (int s4 = 0; s4 < 4; ++s4) {
            const float* p0 = qp + s4 * 32 + hi * 16;
            const float* p1 = qp + s4 * 32 + hi * 16 + 8;
            v4f a0 = *(const v4f*)(p0);
            v4f a1 = *(const v4f*)(p0 + 4);
            v4f b0 = *(const v4f*)(p1);
            v4f b1 = *(const v4f*)(p1 + 4);
            union { _Float16 hh[16]; v16h vv; } u;
            for (int j = 0; j < 4; ++j) {
                u.hh[j]      = (_Float16)a0[j];
                u.hh[4 + j]  = (_Float16)a1[j];
                u.hh[8 + j]  = (_Float16)b0[j];
                u.hh[12 + j] = (_Float16)b1[j];
            }
            qfrag[s4] = u.vv;
        }
    }

    // Online-softmax state. C/D tile layout: lane n = N, VGPR i = row (M = hi*8+i).
    float m_i[8], l_i[8];
    v8f o_acc[8];                      // O strip 16 x 128 as 8 (16x16) f32 tiles
    for (int i = 0; i < 8; ++i) { m_i[i] = -__builtin_inff(); l_i[i] = 0.f; }
    for (int t = 0; t < 8; ++t) o_acc[t] = (v8f){};

    for (int kb = 0; kb <= qb; ++kb) {
        const bool active = (qb - kb < LOCAL) || (((kb + h + 1) % VERT) == 0);
        if (!active) continue;

        __syncthreads();               // previous tile fully consumed
        // ---- stage K (row-major) and V (transposed) tiles as f16 ----------
        {
            const size_t kbase = (size_t)(kb * BLK) * (NKV * HD) + (size_t)hk * HD;
            for (int it = 0; it < 16; ++it) {
                const int li = tid * 4 + it * 512;      // 0..8191, step covers 64x128
                const int r  = li >> 7;
                const int d  = li & 127;
                v4f kv = *(const v4f*)(K + kbase + (size_t)r * (NKV * HD) + d);
                v4f vv = *(const v4f*)(V + kbase + (size_t)r * (NKV * HD) + d);
                _Float16* kd = &Ks[r][d];
                kd[0] = (_Float16)kv[0]; kd[1] = (_Float16)kv[1];
                kd[2] = (_Float16)kv[2]; kd[3] = (_Float16)kv[3];
                Vt[d + 0][r] = (_Float16)vv[0];
                Vt[d + 1][r] = (_Float16)vv[1];
                Vt[d + 2][r] = (_Float16)vv[2];
                Vt[d + 3][r] = (_Float16)vv[3];
            }
        }
        __syncthreads();

        // ---- S = Q * K^T : 4 N-tiles x 4 K-slices = 16 WMMAs --------------
        // B layout (32x16): lane = column n, halves j -> k = hi*16 + j
        // B column n must equal K-row n of this key block => contiguous load.
        v8f sacc[4];
        for (int nt = 0; nt < 4; ++nt) {
            v8f c = (v8f){};
            const int n = nt * 16 + ln16;
            for (int s4 = 0; s4 < 4; ++s4) {
                v16h bfrag = *(const v16h*)&Ks[n][s4 * 32 + hi * 16];
                c = __builtin_amdgcn_wmma_f32_16x16x32_f16(
                        false, qfrag[s4], false, bfrag, (short)0, c, false, false);
            }
            sacc[nt] = c;
        }

        // ---- scale + element-level causal mask on the diagonal block ------
        const bool diag = (kb == qb);
        for (int nt = 0; nt < 4; ++nt) {
            const int kcol = nt * 16 + ln16;            // key idx within block
            for (int i = 0; i < 8; ++i) {
                const int qrow_l = wave * 16 + hi * 8 + i; // query idx within block
                float sv = sacc[nt][i] * ATT_SCALE;
                if (diag && kcol > qrow_l) sv = -1e30f;
                sacc[nt][i] = sv;
            }
        }

        // ---- online softmax (row = fixed VGPR i across one 16-lane half) --
        float pfac[8];
        for (int i = 0; i < 8; ++i) {
            float rm = fmaxf(fmaxf(sacc[0][i], sacc[1][i]),
                             fmaxf(sacc[2][i], sacc[3][i]));
            rm = fmaxf(rm, __shfl_xor(rm, 1, 32));
            rm = fmaxf(rm, __shfl_xor(rm, 2, 32));
            rm = fmaxf(rm, __shfl_xor(rm, 4, 32));
            rm = fmaxf(rm, __shfl_xor(rm, 8, 32));
            const float mnew = fmaxf(m_i[i], rm);
            const float fac  = __expf(m_i[i] - mnew);
            m_i[i] = mnew;
            pfac[i] = fac;
            float rs = 0.f;
            for (int nt = 0; nt < 4; ++nt) {
                float p = __expf(sacc[nt][i] - mnew);
                sacc[nt][i] = p;
                rs += p;
            }
            rs += __shfl_xor(rs, 1, 32);
            rs += __shfl_xor(rs, 2, 32);
            rs += __shfl_xor(rs, 4, 32);
            rs += __shfl_xor(rs, 8, 32);
            l_i[i] = l_i[i] * fac + rs;
        }
        for (int t = 0; t < 8; ++t)
            for (int i = 0; i < 8; ++i)
                o_acc[t][i] *= pfac[i];

        // ---- P (D-layout) -> per-wave LDS -> A-layout fragments -----------
        for (int nt = 0; nt < 4; ++nt) {
            const int col = nt * 16 + ln16;
            for (int i = 0; i < 8; ++i)
                Ps[wave][hi * 8 + i][col] = (_Float16)sacc[nt][i];
        }
        // DS ops from one wave are in-order; compiler inserts s_wait_dscnt
        // before the dependent reads below (same LDS object).
        v16h pfrag[2];
        for (int ks = 0; ks < 2; ++ks) {
            const _Float16* pr = &Ps[wave][ln16][0];
            v8h lo = *(const v8h*)(pr + ks * 32 + hi * 16);
            v8h hv = *(const v8h*)(pr + ks * 32 + hi * 16 + 8);
            pfrag[ks] = __builtin_shufflevector(lo, hv,
                0,1,2,3,4,5,6,7,8,9,10,11,12,13,14,15);
        }

        // ---- O += P * V : 8 dim-tiles x 2 K-slices = 16 WMMAs -------------
        for (int t = 0; t < 8; ++t) {
            const int n = t * 16 + ln16;                 // output dim column
            for (int ks = 0; ks < 2; ++ks) {
                v16h vfrag = *(const v16h*)&Vt[n][ks * 32 + hi * 16];
                o_acc[t] = __builtin_amdgcn_wmma_f32_16x16x32_f16(
                        false, pfrag[ks], false, vfrag, (short)0, o_acc[t],
                        false, false);
            }
        }
    }

    // ---- epilogue: O /= l, store f32 ---------------------------------------
    {
        const int qrow0 = qb * BLK + wave * 16;
        for (int i = 0; i < 8; ++i) {
            const float inv = 1.0f / l_i[i];
            const int qrow = qrow0 + hi * 8 + i;
            float* op = O + (size_t)qrow * (NH * HD) + (size_t)h * HD + ln16;
            for (int t = 0; t < 8; ++t)
                op[t * 16] = o_acc[t][i] * inv;
        }
    }
}

extern "C" void kernel_launch(void* const* d_in, const int* in_sizes, int n_in,
                              void* d_out, int out_size, void* d_ws, size_t ws_size,
                              hipStream_t stream) {
    const float* q = (const float*)d_in[0];   // [2048, 4096]
    const float* k = (const float*)d_in[1];   // [2048, 1024]
    const float* v = (const float*)d_in[2];   // [2048, 1024]
    float* out = (float*)d_out;               // [2048, 4096]
    dim3 grid(NB, NH);                        // 32 query blocks x 32 heads
    dim3 block(128);                          // 4 waves of 32
    bsfa_kernel<<<grid, block, 0, stream>>>(q, k, v, out);
}
